// SequenceGenerator_62603443306541
// MI455X (gfx1250) — matrix operand order repaired and checked by reading, not compiled, not run on hardware
//
#include <hip/hip_runtime.h>
#include <hip/hip_bf16.h>

typedef __attribute__((ext_vector_type(16))) __bf16 v16bf;
typedef __attribute__((ext_vector_type(8)))  __bf16 v8bf;
typedef __attribute__((ext_vector_type(8)))  float  v8f;

#define BATCH 32
#define HS 16
#define WS 16
#define HP 20               /* padded spatial (halo 2 on each side) */
#define WP 20
#define CIN_ 64
#define HID_ 128
#define COUT_ 64
#define TSTEPS 100
#define NPIX (BATCH*HS*WS)          /* 8192 */
#define GCH (4*HID_)                /* 512 */
#define XBP_TOT (BATCH*HP*WP*CIN_)  /* 819200  */
#define HBP_TOT (BATCH*HP*WP*HID_)  /* 1638400 */

// ---------------------------------------------------------------------------
// Weight packing: fp32 HWIO -> bf16 B-fragments in the CDNA5 16-bit B (32x16)
// WMMA lane layout: frag = (tap*numChunks + cc)*numNtiles + ntile, 512 elems;
// element e = lane*16+j : n = ntile*16+(lane&15), k = 8*(lane>>4)+(j&7)+16*(j>>3), c = cc*32+k.
// ---------------------------------------------------------------------------
__global__ void pack_weights_kernel(const float* __restrict__ src,
                                    __bf16* __restrict__ dst,
                                    int numTaps, int numChunks, int Csrc, int Nsrc) {
  int total = numTaps * numChunks * (Nsrc / 16) * 512;
  int idx = blockIdx.x * blockDim.x + threadIdx.x;
  if (idx >= total) return;
  int e      = idx & 511;
  int frag   = idx >> 9;
  int ntiles = Nsrc / 16;
  int ntile  = frag % ntiles;
  int cc     = (frag / ntiles) % numChunks;
  int tap    = frag / (ntiles * numChunks);
  int lane   = e >> 4;
  int j      = e & 15;
  int n = ntile * 16 + (lane & 15);
  int c = cc * 32 + ((lane >> 4) << 3) + (j & 7) + ((j >> 3) << 4);
  dst[idx] = (__bf16)src[((size_t)(tap * Csrc + c)) * Nsrc + n];
}

// Zero padded halos + fill interiors of xb/hb, copy c0 into cbuf.
__global__ void init_state_kernel(const float* __restrict__ x0,
                                  const float* __restrict__ h0,
                                  const float* __restrict__ c0,
                                  __bf16* __restrict__ xbp,
                                  __bf16* __restrict__ hbp,
                                  float* __restrict__ cbuf) {
  int idx = blockIdx.x * blockDim.x + threadIdx.x;
  if (idx < XBP_TOT) {
    int c  = idx % CIN_;
    int r  = idx / CIN_;
    int xx = r % WP, yy = (r / WP) % HP, bi = r / (WP * HP);
    bool in = (yy >= 2 && yy < 18 && xx >= 2 && xx < 18);
    xbp[idx] = in ? (__bf16)x0[((size_t)((bi*HS + yy-2)*WS + xx-2))*CIN_ + c]
                  : (__bf16)0.0f;
  }
  if (idx < HBP_TOT) {
    int c  = idx % HID_;
    int r  = idx / HID_;
    int xx = r % WP, yy = (r / WP) % HP, bi = r / (WP * HP);
    bool in = (yy >= 2 && yy < 18 && xx >= 2 && xx < 18);
    hbp[idx] = in ? (__bf16)h0[((size_t)((bi*HS + yy-2)*WS + xx-2))*HID_ + c]
                  : (__bf16)0.0f;
  }
  if (idx < NPIX * HID_) cbuf[idx] = c0[idx];
}

// Unconditional A-fragment load (padded activations): 16-bit A 16x32 layout.
// p already includes +hi*8 per-lane channel offset; halves j<8 at p[j],
// j>=8 at p[16+j-8].
__device__ __forceinline__ v16bf load_a16(const __bf16* __restrict__ p) {
  v8bf lo  = *(const v8bf*)(p);
  v8bf hi8 = *(const v8bf*)(p + 16);
  v16bf r;
#pragma unroll
  for (int j = 0; j < 8; ++j) { r[j] = lo[j]; r[j + 8] = hi8[j]; }
  return r;
}

// ---------------------------------------------------------------------------
// Gate conv implicit GEMM.  Wave = 16 pixels x 32 channels (2 accumulators,
// shared A fragment).  K-loop: 25 taps x (2 x-chunks + 4 h-chunks), 2 WMMAs
// per K-step -> 300 WMMAs/wave.  grid=(512,4), block=128 (wave -> ntile pair).
// ---------------------------------------------------------------------------
__global__ void gate_gemm_kernel(const __bf16* __restrict__ xbp,
                                 const __bf16* __restrict__ hbp,
                                 const __bf16* __restrict__ WxP,
                                 const __bf16* __restrict__ WhP,
                                 const float* __restrict__ bias,
                                 float* __restrict__ g) {
  int lane   = threadIdx.x & 31;
  int wave   = threadIdx.x >> 5;
  int mt     = blockIdx.x;                 // bi*16 + y
  int npair  = blockIdx.y * 4 + wave;      // 0..15
  int ntile0 = npair * 2;
  int bi = mt >> 4, y = mt & 15;
  int m = lane & 15, hi = lane >> 4;

  // Per-lane base pointers into padded activations (tap (0,0) position).
  const __bf16* xbase = xbp + ((size_t)((bi*HP + y)*WP + m))*CIN_ + hi*8;
  const __bf16* hbase = hbp + ((size_t)((bi*HP + y)*WP + m))*HID_ + hi*8;

  v8f acc0 = {}, acc1 = {};
  for (int tap = 0; tap < 25; ++tap) {
    int dy = tap / 5, dx = tap % 5;
    const __bf16* xa = xbase + (size_t)(dy*WP + dx)*CIN_;
    const __bf16* ha = hbase + (size_t)(dy*WP + dx)*HID_;
#pragma unroll
    for (int cc = 0; cc < 2; ++cc) {       // x contribution (CIN=64)
      v16bf a = load_a16(xa + cc*32);
      const __bf16* bp = WxP + ((size_t)((tap*2 + cc)*32 + ntile0))*512;
      v16bf b0 = *(const v16bf*)(bp + lane*16);
      v16bf b1 = *(const v16bf*)(bp + 512 + lane*16);
      acc0 = __builtin_amdgcn_wmma_f32_16x16x32_bf16(false, a, false, b0, (short)0, acc0, false, false);
      acc1 = __builtin_amdgcn_wmma_f32_16x16x32_bf16(false, a, false, b1, (short)0, acc1, false, false);
    }
#pragma unroll
    for (int cc = 0; cc < 4; ++cc) {       // h contribution (HID=128)
      v16bf a = load_a16(ha + cc*32);
      const __bf16* bp = WhP + ((size_t)((tap*4 + cc)*32 + ntile0))*512;
      v16bf b0 = *(const v16bf*)(bp + lane*16);
      v16bf b1 = *(const v16bf*)(bp + 512 + lane*16);
      acc0 = __builtin_amdgcn_wmma_f32_16x16x32_bf16(false, a, false, b0, (short)0, acc0, false, false);
      acc1 = __builtin_amdgcn_wmma_f32_16x16x32_bf16(false, a, false, b1, (short)0, acc1, false, false);
    }
  }
  // C/D layout: VGPR r -> M = r + 8*(lane>>4), N = lane&15.
  int n = lane & 15;
  size_t pixBase = (size_t)mt * 16;
  float bv0 = bias[ntile0*16 + n];
  float bv1 = bias[(ntile0+1)*16 + n];
#pragma unroll
  for (int r = 0; r < 8; ++r) {
    int mm = r + hi*8;
    g[(pixBase + mm)*GCH + ntile0*16 + n]     = acc0[r] + bv0;
    g[(pixBase + mm)*GCH + (ntile0+1)*16 + n] = acc1[r] + bv1;
  }
}

// LSTM pointwise update; writes h (bf16) into padded interior.
__global__ void lstm_update_kernel(const float* __restrict__ g,
                                   float* __restrict__ cbuf,
                                   __bf16* __restrict__ hbp) {
  int idx = blockIdx.x * blockDim.x + threadIdx.x;
  if (idx >= NPIX * HID_) return;
  int p = idx / HID_, ch = idx % HID_;
  const float* gp = g + (size_t)p * GCH;
  float ig = gp[ch];
  float fg = gp[HID_ + ch];
  float cg = gp[2*HID_ + ch];
  float og = gp[3*HID_ + ch];
  float i_s = 1.f / (1.f + __expf(-ig));
  float f_s = 1.f / (1.f + __expf(-fg));
  float o_s = 1.f / (1.f + __expf(-og));
  float cn = f_s * cbuf[idx] + i_s * tanhf(cg);
  cbuf[idx] = cn;
  int bi = p >> 8, y = (p >> 4) & 15, x = p & 15;
  hbp[((size_t)((bi*HP + y+2)*WP + x+2))*HID_ + ch] = (__bf16)(o_s * tanhf(cn));
}

// Out conv 3x3 (HID->COUT).  Wave = 16 pixels x 32 channels (2 accumulators).
// grid=(512), block=64 (2 waves -> ntile pairs {0,1},{2,3}).  72 WMMAs/wave.
__global__ void out_gemm_kernel(const __bf16* __restrict__ hbp,
                                const __bf16* __restrict__ CwP,
                                const float* __restrict__ bias,
                                float* __restrict__ ybuf) {
  int lane   = threadIdx.x & 31;
  int ntile0 = (threadIdx.x >> 5) * 2;     // 0 or 2
  int mt     = blockIdx.x;
  int bi = mt >> 4, y = mt & 15;
  int m = lane & 15, hi = lane >> 4;

  // 3x3 conv with pad 1 inside a halo-2 buffer: tap (dy,dx) -> row y+dy+1.
  const __bf16* hbase = hbp + ((size_t)((bi*HP + y+1)*WP + m+1))*HID_ + hi*8;

  v8f acc0 = {}, acc1 = {};
  for (int tap = 0; tap < 9; ++tap) {
    int dy = tap / 3, dx = tap % 3;
    const __bf16* ha = hbase + (size_t)(dy*WP + dx)*HID_;
#pragma unroll
    for (int cc = 0; cc < 4; ++cc) {
      v16bf a = load_a16(ha + cc*32);
      const __bf16* bp = CwP + ((size_t)((tap*4 + cc)*4 + ntile0))*512;
      v16bf b0 = *(const v16bf*)(bp + lane*16);
      v16bf b1 = *(const v16bf*)(bp + 512 + lane*16);
      acc0 = __builtin_amdgcn_wmma_f32_16x16x32_bf16(false, a, false, b0, (short)0, acc0, false, false);
      acc1 = __builtin_amdgcn_wmma_f32_16x16x32_bf16(false, a, false, b1, (short)0, acc1, false, false);
    }
  }
  int n = lane & 15;
  size_t pixBase = (size_t)mt * 16;
  float bv0 = bias[ntile0*16 + n];
  float bv1 = bias[(ntile0+1)*16 + n];
#pragma unroll
  for (int r = 0; r < 8; ++r) {
    int mm = r + hi*8;
    ybuf[(pixBase + mm)*COUT_ + ntile0*16 + n]     = acc0[r] + bv0;
    ybuf[(pixBase + mm)*COUT_ + (ntile0+1)*16 + n] = acc1[r] + bv1;
  }
}

// LayerNorm(64) + LeakyReLU + write out[:,t] + autoregressive next-x (padded).
__global__ void norm_out_kernel(const float* __restrict__ ybuf,
                                const float* __restrict__ gamma,
                                const float* __restrict__ beta,
                                const float* __restrict__ targets,
                                const float* __restrict__ autoreg_prob,
                                float* __restrict__ out,
                                __bf16* __restrict__ xbp,
                                int t) {
  int p = blockIdx.x * blockDim.x + threadIdx.x;
  if (p >= NPIX) return;
  const float* yp = ybuf + (size_t)p * COUT_;
  float s = 0.f, s2 = 0.f;
#pragma unroll 8
  for (int ch = 0; ch < COUT_; ++ch) { float v = yp[ch]; s += v; s2 += v * v; }
  float mu  = s * (1.f / COUT_);
  float var = s2 * (1.f / COUT_) - mu * mu;
  float rs  = rsqrtf(var + 1e-3f);
  int bi = p >> 8, hw = p & 255;
  int y = hw >> 4, x = hw & 15;
  size_t obase = ((size_t)(bi * TSTEPS + t) * (HS * WS) + hw) * COUT_;
  size_t xbase = ((size_t)((bi*HP + y+2)*WP + x+2)) * CIN_;
  bool auto_reg = (float)t < (float)TSTEPS * autoreg_prob[0];
#pragma unroll 4
  for (int ch = 0; ch < COUT_; ++ch) {
    float v = (yp[ch] - mu) * rs * gamma[ch] + beta[ch];
    float o = v >= 0.f ? v : 0.2f * v;
    out[obase + ch] = o;
    float nx = auto_reg ? o : targets[obase + ch];
    xbp[xbase + ch] = (__bf16)nx;
  }
}

// ---------------------------------------------------------------------------
static inline char* align256(char* p) {
  return (char*)(((uintptr_t)p + 255) & ~(uintptr_t)255);
}

extern "C" void kernel_launch(void* const* d_in, const int* in_sizes, int n_in,
                              void* d_out, int out_size, void* d_ws, size_t ws_size,
                              hipStream_t stream) {
  const float* initial_input = (const float*)d_in[0];
  const float* h0            = (const float*)d_in[1];
  const float* c0            = (const float*)d_in[2];
  const float* targets       = (const float*)d_in[3];
  const float* autoreg_prob  = (const float*)d_in[4];
  const float* Wx            = (const float*)d_in[5];
  const float* Wh            = (const float*)d_in[6];
  const float* bias_g        = (const float*)d_in[7];
  const float* conv_w        = (const float*)d_in[8];
  const float* conv_b        = (const float*)d_in[9];
  const float* gamma         = (const float*)d_in[10];
  const float* beta          = (const float*)d_in[11];
  float* out = (float*)d_out;

  char* p = (char*)d_ws;
  p = align256(p); __bf16* WxP  = (__bf16*)p; p += (size_t)25 * 2 * 32 * 512 * 2;
  p = align256(p); __bf16* WhP  = (__bf16*)p; p += (size_t)25 * 4 * 32 * 512 * 2;
  p = align256(p); __bf16* CwP  = (__bf16*)p; p += (size_t)9  * 4 * 4  * 512 * 2;
  p = align256(p); __bf16* xbp  = (__bf16*)p; p += (size_t)XBP_TOT * 2;
  p = align256(p); __bf16* hbp  = (__bf16*)p; p += (size_t)HBP_TOT * 2;
  p = align256(p); float*  cbuf = (float*)p;  p += (size_t)NPIX * HID_ * 4;
  p = align256(p); float*  gbuf = (float*)p;  p += (size_t)NPIX * GCH  * 4;
  p = align256(p); float*  ybuf = (float*)p;  p += (size_t)NPIX * COUT_ * 4;
  (void)ws_size;

  {
    int totWx = 25 * 2 * 32 * 512;
    pack_weights_kernel<<<(totWx + 255) / 256, 256, 0, stream>>>(Wx, WxP, 25, 2, CIN_, GCH);
    int totWh = 25 * 4 * 32 * 512;
    pack_weights_kernel<<<(totWh + 255) / 256, 256, 0, stream>>>(Wh, WhP, 25, 4, HID_, GCH);
    int totCw = 9 * 4 * 4 * 512;
    pack_weights_kernel<<<(totCw + 255) / 256, 256, 0, stream>>>(conv_w, CwP, 9, 4, HID_, COUT_);
  }
  init_state_kernel<<<(HBP_TOT + 255) / 256, 256, 0, stream>>>(initial_input, h0, c0, xbp, hbp, cbuf);

  dim3 gateGrid(BATCH * HS, 4);   // 512 M-tiles x (4 blocks * 4 waves = 16 ntile pairs)
  dim3 gateBlock(128);
  dim3 outGrid(BATCH * HS);
  dim3 outBlock(64);              // 2 waves -> 2 ntile pairs

  for (int t = 0; t < TSTEPS; ++t) {
    gate_gemm_kernel<<<gateGrid, gateBlock, 0, stream>>>(xbp, hbp, WxP, WhP, bias_g, gbuf);
    lstm_update_kernel<<<(NPIX * HID_ + 255) / 256, 256, 0, stream>>>(gbuf, cbuf, hbp);
    out_gemm_kernel<<<outGrid, outBlock, 0, stream>>>(hbp, CwP, conv_b, ybuf);
    norm_out_kernel<<<(NPIX + 255) / 256, 256, 0, stream>>>(ybuf, gamma, beta, targets,
                                                            autoreg_prob, out, xbp, t);
  }
  (void)in_sizes; (void)n_in; (void)out_size;
}
